// PlaneTransformer_34703335752240
// MI455X (gfx1250) — matrix operand order repaired and checked by reading, not compiled, not run on hardware
//
#include <hip/hip_runtime.h>
#include <stdint.h>

typedef __bf16 bf16_t;
typedef __attribute__((ext_vector_type(16))) __bf16 v16bf;
typedef __attribute__((ext_vector_type(8)))  __bf16 v8bf;
typedef __attribute__((ext_vector_type(8)))  float  v8f;

#define DEV __device__ __forceinline__

DEV bf16_t f2bf(float f){
  union { float f; uint32_t u; } v; v.f = f;
  uint32_t u = v.u;
  u += 0x7fffu + ((u >> 16) & 1u);            // round-to-nearest-even
  uint16_t b = (uint16_t)(u >> 16);
  bf16_t r; __builtin_memcpy(&r, &b, 2); return r;
}
DEV float bf2f(bf16_t h){
  uint16_t b; __builtin_memcpy(&b, &h, 2);
  union { uint32_t u; float f; } v; v.u = ((uint32_t)b) << 16; return v.f;
}

// A fragment (16x32 bf16, row-major source): lane m = lane&15; K chunks
// half*8..+7 (VGPR0-3) and 16+half*8..+7 (VGPR4-7), packed 2/VGPR.
DEV v16bf frag_a(const bf16_t* row, int half){
  v8bf lo = *(const v8bf*)(row + half*8);
  v8bf hi = *(const v8bf*)(row + 16 + half*8);
  v16bf r;
#pragma unroll
  for (int i=0;i<8;++i){ r[i]=lo[i]; r[i+8]=hi[i]; }
  return r;
}
// B fragment (32x16 bf16) from COLUMN-MAJOR storage ([N][K] rows): lane
// n = lane&15; K chunks half*8..+7 and 16+half*8..+7 (same as A).
DEV v16bf frag_b(const bf16_t* rowN, int half){
  v8bf lo = *(const v8bf*)(rowN + half*8);
  v8bf hi = *(const v8bf*)(rowN + 16 + half*8);
  v16bf r;
#pragma unroll
  for (int i=0;i<8;++i){ r[i]=lo[i]; r[i+8]=hi[i]; }
  return r;
}
DEV v8f wmma_bf(v16bf a, v16bf b, v8f c){
  return __builtin_amdgcn_wmma_f32_16x16x32_bf16(false, a, false, b, (short)0, c, false, false);
}

// ---------------------------------------------------------------- constants
#define CC   256
#define SD   32
#define NPIX 1024          // H*W
#define BT   64            // B*D
#define NROW 65536         // BT*NPIX  (also B*32^3 conv positions)

// ------------------------------------------------------- 1. transpose + LN
__global__ void k_ln(const float* __restrict__ x, const float* __restrict__ lnw,
                     const float* __restrict__ lnb, bf16_t* __restrict__ xn)
{
  int bid = blockIdx.x;                 // 2048 = B*H*W
  int b  = bid >> 10, hh = (bid >> 5) & 31, ww = bid & 31;
  __shared__ float tile[CC][SD];        // [c][d]
  __shared__ float part[2][8][SD];
  __shared__ float mrow[SD], rrow[SD];
  int t = threadIdx.x;
  int d = t & 31, cg = t >> 5;          // cg 0..7
  for (int j = 0; j < 32; ++j) {
    int c = cg*32 + j;
    tile[c][d] = x[(((size_t)(b*CC + c)*SD + hh)*SD + ww)*SD + d];
  }
  __syncthreads();
  float s = 0.f, s2 = 0.f;
  for (int j = 0; j < 32; ++j){ float v = tile[cg*32+j][d]; s += v; s2 += v*v; }
  part[0][cg][d] = s; part[1][cg][d] = s2;
  __syncthreads();
  if (t < 32) {
    float a = 0.f, a2 = 0.f;
    for (int g = 0; g < 8; ++g){ a += part[0][g][t]; a2 += part[1][g][t]; }
    float m = a * (1.f/256.f);
    float var = a2 * (1.f/256.f) - m*m;
    mrow[t] = m; rrow[t] = rsqrtf(var + 1e-5f);
  }
  __syncthreads();
  int n = hh*32 + ww;
  for (int j = 0; j < 32; ++j) {
    int c = cg*32 + j;
    float v = (tile[c][d] - mrow[d]) * rrow[d] * lnw[c] + lnb[c];
    xn[((size_t)(b*32 + d)*NPIX + n)*CC + c] = f2bf(v);
  }
}

// ------------------------------------------------------------ weight casts
__global__ void k_castT(const float* __restrict__ in, bf16_t* __restrict__ out,
                        int R, int Cl){
  int i = blockIdx.x*256 + threadIdx.x;
  if (i < R*Cl){ int r = i / Cl, c = i % Cl; out[(size_t)c*R + r] = f2bf(in[i]); }
}
__global__ void k_castConvW(const float* __restrict__ in, bf16_t* __restrict__ out){
  int i = blockIdx.x*256 + threadIdx.x;               // [Co][Ci][27]->[27][Co][Ci]
  if (i < 256*256*27){
    int co = i / (256*27); int rem = i % (256*27); int ci = rem / 27; int tap = rem % 27;
    out[((size_t)tap*256 + co)*256 + ci] = f2bf(in[i]);
  }
}
__global__ void k_cast(const float* __restrict__ in, bf16_t* __restrict__ out, int n){
  int i = blockIdx.x*256 + threadIdx.x;
  if (i < n) out[i] = f2bf(in[i]);
}
__global__ void k_zero(float* __restrict__ p, int n){
  int i = blockIdx.x*256 + threadIdx.x;
  if (i < n) p[i] = 0.f;
}

// ----------------------------------------------------------- 3. QKVV GEMM
// xn [65536][256] @ qkvv_w -> qkvv [65536][1024]; one wave = 16 rows x 64 cols
__global__ void k_gemm_qkvv(const bf16_t* __restrict__ A,
                            const bf16_t* __restrict__ Bt_,   // [1024][256] col-major
                            bf16_t* __restrict__ Dst)
{
  int gw = (blockIdx.x*blockDim.x + threadIdx.x) >> 5;
  int lane = threadIdx.x & 31, half = lane >> 4, nl = lane & 15;
  int mt = gw >> 4, ng = gw & 15;
  int m0 = mt * 16;
  v8f acc[4] = {};
  for (int k0 = 0; k0 < 256; k0 += 32) {
    if (k0 + 32 < 256) {            // pull next K-chunk a step ahead
      __builtin_prefetch(A   + (size_t)(m0 + nl)*256 + k0 + 32, 0, 1);
      __builtin_prefetch(Bt_ + (size_t)(ng*64 + lane)*256 + k0 + 32, 0, 1);
      __builtin_prefetch(Bt_ + (size_t)(ng*64 + 32 + lane)*256 + k0 + 32, 0, 1);
    }
    v16bf af = frag_a(A + (size_t)(m0 + nl)*256 + k0, half);
#pragma unroll
    for (int nt = 0; nt < 4; ++nt)
      acc[nt] = wmma_bf(af, frag_b(Bt_ + (size_t)(ng*64 + nt*16 + nl)*256 + k0, half), acc[nt]);
  }
#pragma unroll
  for (int nt = 0; nt < 4; ++nt)
#pragma unroll
    for (int r = 0; r < 8; ++r)
      Dst[(size_t)(m0 + r + 8*half)*1024 + ng*64 + nt*16 + nl] = f2bf(acc[nt][r]);
}

// ------------------- 4. q/k column inverse L2 norms (zero -> partial -> fin)
__global__ void k_invpart(const bf16_t* __restrict__ qkvv, float* __restrict__ acc){
  int bid = blockIdx.x;                 // 1024 = 64 bt x 16 row-chunks
  int bt = bid >> 4, rc = bid & 15;
  int n0 = rc * 64;
  for (int cc = threadIdx.x; cc < 512; cc += 256){
    const bf16_t* p = qkvv + (size_t)bt*NPIX*1024 + (size_t)n0*1024 + cc;
    float s = 0.f;
    for (int n = 0; n < 64; ++n){ float v = bf2f(p[(size_t)n*1024]); s += v*v; }
    atomicAdd(&acc[bt*512 + cc], s);
  }
}
__global__ void k_invfin(float* __restrict__ a, int n){
  int i = blockIdx.x*256 + threadIdx.x;
  if (i < n) a[i] = 1.f / fmaxf(sqrtf(a[i]), 1e-12f);
}

// ------------------------------------------- 5. per-(slice,head) attention
__global__ void __launch_bounds__(128)
k_attn(const bf16_t* __restrict__ qkvv, const bf16_t* __restrict__ EwT,
       const float* __restrict__ E_b, const float* __restrict__ inv,
       const float* __restrict__ temp1, const float* __restrict__ temp2,
       bf16_t* __restrict__ xca, bf16_t* __restrict__ xsa)
{
  int bid = blockIdx.x, bt = bid >> 3, h = bid & 7;
  int t = threadIdx.x, wv = t >> 5, lane = t & 31;
  int half = lane >> 4, nl = lane & 15;
  size_t qbase = (size_t)bt*NPIX*1024;
  int qoff = h*32, koff = 256 + h*32, vcoff = 512 + h*32, vsoff = 768 + h*32;

  __shared__ __align__(16) bf16_t Qt[32][40], Kt[32][40], Vt[32][40];  // transposed chunk
  __shared__ __align__(16) float   sca[32][32];
  __shared__ __align__(16) bf16_t  attnca[32][32];
  __shared__ __align__(16) bf16_t  kpT[64][32];   // (kp + E_b) * invq * temp2, [p][dc]
  __shared__ __align__(16) bf16_t  vpL[32][64];   // vp + E_b, [dc][p]
  __shared__ __align__(16) float   scf[4][16][64];
  __shared__ __align__(16) bf16_t  asa[4][16][64];

  v8f accS[4]  = {};   // wave0: Q^T K   (32x32)
  v8f accKP[8] = {};   // wave1: K^T E   (32x64)
  v8f accVP[8] = {};   // wave2: Vsa^T E (32x64)

  // ---- phase alpha: reductions over N
  for (int n0 = 0; n0 < NPIX; n0 += 32) {
    int nloc = t >> 2, dc0 = (t & 3) * 8;
    const bf16_t* src = qkvv + qbase + (size_t)(n0 + nloc)*1024;
    if (n0 + 32 < NPIX)
      __builtin_prefetch(qkvv + qbase + (size_t)(n0 + 32 + nloc)*1024 + qoff, 0, 1);
    v8bf q = *(const v8bf*)(src + qoff + dc0);
    v8bf k = *(const v8bf*)(src + koff + dc0);
    v8bf v = *(const v8bf*)(src + vsoff + dc0);
#pragma unroll
    for (int i = 0; i < 8; ++i){ Qt[dc0+i][nloc]=q[i]; Kt[dc0+i][nloc]=k[i]; Vt[dc0+i][nloc]=v[i]; }
    __syncthreads();
    if (wv == 0) {
#pragma unroll
      for (int mt2 = 0; mt2 < 2; ++mt2){
        v16bf a = frag_a(&Qt[mt2*16 + nl][0], half);
#pragma unroll
        for (int nt2 = 0; nt2 < 2; ++nt2)
          accS[mt2*2+nt2] = wmma_bf(a, frag_b(&Kt[nt2*16 + nl][0], half), accS[mt2*2+nt2]);
      }
    } else if (wv == 1) {
#pragma unroll
      for (int mt2 = 0; mt2 < 2; ++mt2){
        v16bf a = frag_a(&Kt[mt2*16 + nl][0], half);
#pragma unroll
        for (int nt2 = 0; nt2 < 4; ++nt2)
          accKP[mt2*4+nt2] = wmma_bf(a, frag_b(EwT + (size_t)(nt2*16 + nl)*1024 + n0, half), accKP[mt2*4+nt2]);
      }
    } else if (wv == 2) {
#pragma unroll
      for (int mt2 = 0; mt2 < 2; ++mt2){
        v16bf a = frag_a(&Vt[mt2*16 + nl][0], half);
#pragma unroll
        for (int nt2 = 0; nt2 < 4; ++nt2)
          accVP[mt2*4+nt2] = wmma_bf(a, frag_b(EwT + (size_t)(nt2*16 + nl)*1024 + n0, half), accVP[mt2*4+nt2]);
      }
    }
    __syncthreads();
  }

  float tc = temp1[h], ts = temp2[h];
  const float* invq = inv + bt*512 + h*32;
  const float* invk = inv + bt*512 + 256 + h*32;
  if (wv == 0) {
#pragma unroll
    for (int mt2 = 0; mt2 < 2; ++mt2)
#pragma unroll
      for (int nt2 = 0; nt2 < 2; ++nt2)
#pragma unroll
        for (int r = 0; r < 8; ++r){
          int dcq = mt2*16 + r + 8*half, e = nt2*16 + nl;
          sca[dcq][e] = accS[mt2*2+nt2][r] * invq[dcq] * invk[e] * tc;
        }
  } else if (wv == 1) {
#pragma unroll
    for (int mt2 = 0; mt2 < 2; ++mt2)
#pragma unroll
      for (int nt2 = 0; nt2 < 4; ++nt2)
#pragma unroll
        for (int r = 0; r < 8; ++r){
          int dc = mt2*16 + r + 8*half, p = nt2*16 + nl;
          kpT[p][dc] = f2bf((accKP[mt2*4+nt2][r] + E_b[p]) * invq[dc] * ts);
        }
  } else if (wv == 2) {
#pragma unroll
    for (int mt2 = 0; mt2 < 2; ++mt2)
#pragma unroll
      for (int nt2 = 0; nt2 < 4; ++nt2)
#pragma unroll
        for (int r = 0; r < 8; ++r){
          int dc = mt2*16 + r + 8*half, p = nt2*16 + nl;
          vpL[dc][p] = f2bf(accVP[mt2*4+nt2][r] + E_b[p]);
        }
  }
  __syncthreads();
  if (t < 32) {                       // channel-attn softmax (rows over e)
    float mx = -1e30f;
    for (int e = 0; e < 32; ++e) mx = fmaxf(mx, sca[t][e]);
    float s = 0.f;
    for (int e = 0; e < 32; ++e){ float ev = __expf(sca[t][e]-mx); s += ev; sca[t][e] = ev; }
    float rs = 1.f / s;
    for (int e = 0; e < 32; ++e) attnca[t][e] = f2bf(sca[t][e]*rs);
  }
  __syncthreads();

  // ---- phase beta: stream over N tiles (each wave owns 16 rows)
  for (int n0 = 0; n0 < NPIX; n0 += 64) {
    int m0 = n0 + wv*16;
    if (n0 + 64 < NPIX)
      __builtin_prefetch(qkvv + qbase + (size_t)(n0 + 64 + wv*16 + nl)*1024 + qoff, 0, 1);
    { // x_ca = Vca[16x32] @ attnca^T   (K=32)
      v16bf a = frag_a(qkvv + qbase + (size_t)(m0 + nl)*1024 + vcoff, half);
#pragma unroll
      for (int nt2 = 0; nt2 < 2; ++nt2){
        v8f acc = {};
        acc = wmma_bf(a, frag_b(&attnca[nt2*16 + nl][0], half), acc);
#pragma unroll
        for (int r = 0; r < 8; ++r){
          int n = m0 + r + 8*half, dc = nt2*16 + nl;
          xca[((size_t)bt*NPIX + n)*CC + h*32 + dc] = f2bf(acc[r]);
        }
      }
    }
    { // scores = Q[16x32] @ kp' [32x64]
      v16bf a = frag_a(qkvv + qbase + (size_t)(m0 + nl)*1024 + qoff, half);
#pragma unroll
      for (int nt2 = 0; nt2 < 4; ++nt2){
        v8f z = {};
        z = wmma_bf(a, frag_b(&kpT[nt2*16 + nl][0], half), z);
#pragma unroll
        for (int r = 0; r < 8; ++r) scf[wv][r + 8*half][nt2*16 + nl] = z[r];
      }
    }
    __syncthreads();
    if (lane < 16) {                  // spatial softmax over P=64
      float mx = -1e30f;
      for (int p = 0; p < 64; ++p) mx = fmaxf(mx, scf[wv][lane][p]);
      float s = 0.f;
      for (int p = 0; p < 64; ++p){ float ev = __expf(scf[wv][lane][p]-mx); s += ev; scf[wv][lane][p] = ev; }
      float rs = 1.f / s;
      for (int p = 0; p < 64; ++p) asa[wv][lane][p] = f2bf(scf[wv][lane][p]*rs);
    }
    __syncthreads();
    { // x_sa = attn_sa[16x64] @ vp^T[64x32]
      v8f acc2[2] = {};
#pragma unroll
      for (int kc = 0; kc < 2; ++kc){
        v16bf a = frag_a(&asa[wv][nl][kc*32], half);
#pragma unroll
        for (int nt2 = 0; nt2 < 2; ++nt2)
          acc2[nt2] = wmma_bf(a, frag_b(&vpL[nt2*16 + nl][kc*32], half), acc2[nt2]);
      }
#pragma unroll
      for (int nt2 = 0; nt2 < 2; ++nt2)
#pragma unroll
        for (int r = 0; r < 8; ++r){
          int n = m0 + r + 8*half, dc = nt2*16 + nl;
          int np = dc*32 + h*4 + (n >> 8);    // faithful permute+flat reshape
          int cp = n & 255;
          xsa[((size_t)bt*NPIX + np)*CC + cp] = f2bf(acc2[nt2][r]);
        }
    }
    __syncthreads();
  }
}

// --------------------------------- 6. out projections + gamma residual
__global__ void k_proj(const bf16_t* __restrict__ xsa, const bf16_t* __restrict__ xca,
                       const bf16_t* __restrict__ W1t, const bf16_t* __restrict__ W2t,
                       const float* __restrict__ b1, const float* __restrict__ b2,
                       const float* __restrict__ gamma, const float* __restrict__ x,
                       float* __restrict__ askip, bf16_t* __restrict__ act0)
{
  int gw = (blockIdx.x*blockDim.x + threadIdx.x) >> 5;     // 8192 waves
  int lane = threadIdx.x & 31, half = lane >> 4, nl = lane & 15;
  int branch = gw & 1, mt = gw >> 1;
  const bf16_t* Am = branch ? xca : xsa;
  const bf16_t* Wt = branch ? W2t : W1t;
  const float*  bb = branch ? b2  : b1;
  int m0 = mt * 16;
  v8f acc[8] = {};
  for (int k0 = 0; k0 < 256; k0 += 32) {
    if (k0 + 32 < 256)
      __builtin_prefetch(Am + (size_t)(m0 + nl)*256 + k0 + 32, 0, 1);
    v16bf a = frag_a(Am + (size_t)(m0 + nl)*256 + k0, half);
#pragma unroll
    for (int nt2 = 0; nt2 < 8; ++nt2)
      acc[nt2] = wmma_bf(a, frag_b(Wt + (size_t)(nt2*16 + nl)*256 + k0, half), acc[nt2]);
  }
#pragma unroll
  for (int nt2 = 0; nt2 < 8; ++nt2)
#pragma unroll
    for (int r = 0; r < 8; ++r){
      int row = m0 + r + 8*half;
      int cl = nt2*16 + nl, c = branch*128 + cl;
      int bt = row >> 10, n = row & 1023;
      int b = bt >> 5, d = bt & 31, hh = n >> 5, ww = n & 31;
      size_t xi = (((size_t)(b*CC + c)*SD + hh)*SD + ww)*SD + d;
      float v = x[xi] + gamma[c]*(acc[nt2][r] + bb[cl]);
      askip[xi] = v;
      act0[((size_t)((b*32 + hh)*32 + ww)*32 + d)*CC + c] = f2bf(v);
    }
}

// --------------------------- 7. 3x3x3 conv as implicit GEMM (channels-last)
__global__ void k_conv(const bf16_t* __restrict__ act, const bf16_t* __restrict__ wt,
                       float* __restrict__ outp)
{
  int bid = blockIdx.x;                     // 2048 = B*32*32, row tile = 32 x-positions
  int b = bid >> 10, z = (bid >> 5) & 31, y = bid & 31;
  int t = threadIdx.x, wv = t >> 5, lane = t & 31, half = lane >> 4, nl = lane & 15;
  __shared__ __align__(16) bf16_t stg[34][256];
  v8f acc[4] = {};
  for (int dz = -1; dz <= 1; ++dz){
    int zz = z + dz; if (zz < 0 || zz > 31) continue;
    for (int dy = -1; dy <= 1; ++dy){
      int yy = y + dy; if (yy < 0 || yy > 31) continue;
      __syncthreads();
      for (int i = t; i < 34*32; i += 256){
        int row = i >> 5, seg = i & 31, xx = row - 1;
        if (xx < 0 || xx > 31){
          for (int e = 0; e < 8; ++e) stg[row][seg*8+e] = f2bf(0.f);
        } else {
          v8bf vv = *(const v8bf*)(act + (((size_t)(b*32 + zz)*32 + yy)*32 + xx)*CC + seg*8);
          for (int e = 0; e < 8; ++e) stg[row][seg*8+e] = vv[e];
        }
      }
      // prefetch the next y-plane of activations + next tap group's weights
      if (yy + 1 <= 31)
        __builtin_prefetch(act + (((size_t)(b*32 + zz)*32 + (yy+1))*32 + (t & 31))*CC, 0, 1);
      {
        int tapb = (dz+1)*9 + (dy+1)*3;
        if (tapb + 3 < 27)
          __builtin_prefetch(wt + ((size_t)(tapb+3)*256 + wv*32 + lane)*256, 0, 1);
      }
      __syncthreads();
      for (int dx = 0; dx < 3; ++dx){
        int tap = (dz+1)*9 + (dy+1)*3 + dx;
        const bf16_t* wtap = wt + (size_t)tap*256*256;
        for (int k0 = 0; k0 < 256; k0 += 32){
#pragma unroll
          for (int mt2 = 0; mt2 < 2; ++mt2){
            v16bf a = frag_a(&stg[mt2*16 + nl + dx][k0], half);
#pragma unroll
            for (int nt2 = 0; nt2 < 2; ++nt2)
              acc[mt2*2+nt2] = wmma_bf(a,
                  frag_b(wtap + (size_t)(wv*32 + nt2*16 + nl)*256 + k0, half),
                  acc[mt2*2+nt2]);
          }
        }
      }
    }
  }
#pragma unroll
  for (int mt2 = 0; mt2 < 2; ++mt2)
#pragma unroll
    for (int nt2 = 0; nt2 < 2; ++nt2)
#pragma unroll
      for (int r = 0; r < 8; ++r){
        int xx = mt2*16 + r + 8*half, co = wv*32 + nt2*16 + nl;
        outp[(((size_t)(b*32 + z)*32 + y)*32 + xx)*CC + co] = acc[mt2*2+nt2][r];
      }
}

// ------------------------------------------- 8. instance-norm stats + apply
__global__ void k_instat(const float* __restrict__ co_, float* __restrict__ stats){
  int bc = blockIdx.x, b = bc >> 8, c = bc & 255;
  __shared__ float sh[2][256];
  float s = 0.f, s2 = 0.f;
  for (int i = threadIdx.x; i < 32768; i += 256){
    float v = co_[((size_t)b*32768 + i)*CC + c];
    s += v; s2 += v*v;
  }
  sh[0][threadIdx.x] = s; sh[1][threadIdx.x] = s2;
  __syncthreads();
  for (int st = 128; st > 0; st >>= 1){
    if (threadIdx.x < st){ sh[0][threadIdx.x] += sh[0][threadIdx.x+st];
                           sh[1][threadIdx.x] += sh[1][threadIdx.x+st]; }
    __syncthreads();
  }
  if (threadIdx.x == 0){
    float m = sh[0][0] * (1.f/32768.f);
    float var = sh[1][0] * (1.f/32768.f) - m*m;
    stats[bc*2] = m; stats[bc*2+1] = rsqrtf(var + 1e-5f);
  }
}

__global__ void k_inapply(const float* __restrict__ co_, const float* __restrict__ stats,
                          const float* __restrict__ skip, bf16_t* __restrict__ out, int mode){
  size_t i = (size_t)blockIdx.x*256 + threadIdx.x;
  if (i >= (size_t)16777216) return;
  int c = (int)(i & 255);
  size_t p = i >> 8;
  int b = (int)(p >> 15), rem = (int)(p & 32767);
  float m = stats[(b*256+c)*2], rs = stats[(b*256+c)*2+1];
  float v = (co_[i] - m) * rs;
  if (mode == 2){
    int z = rem >> 10, y = (rem >> 5) & 31, xx = rem & 31;
    v += skip[(((size_t)(b*CC + c)*SD + z)*SD + y)*SD + xx];
  }
  v = v > 0.f ? v : 0.01f*v;
  out[i] = f2bf(v);
}

// ------------------------------------ 9. 1x1x1 conv + final residual to out
__global__ void k_conv8(const bf16_t* __restrict__ act2, const bf16_t* __restrict__ w8,
                        const float* __restrict__ b8, const float* __restrict__ skip,
                        float* __restrict__ outp){
  int gw = (blockIdx.x*blockDim.x + threadIdx.x) >> 5;   // 16384 waves
  int lane = threadIdx.x & 31, half = lane >> 4, nl = lane & 15;
  int mt = gw >> 2, ng = gw & 3;
  int m0 = mt * 16;
  v8f acc[4] = {};
  for (int k0 = 0; k0 < 256; k0 += 32) {
    if (k0 + 32 < 256) {
      __builtin_prefetch(act2 + (size_t)(m0 + nl)*256 + k0 + 32, 0, 1);
      __builtin_prefetch(w8 + (size_t)(ng*64 + lane)*256 + k0 + 32, 0, 1);
    }
    v16bf a = frag_a(act2 + (size_t)(m0 + nl)*256 + k0, half);
#pragma unroll
    for (int nt2 = 0; nt2 < 4; ++nt2)
      acc[nt2] = wmma_bf(a, frag_b(w8 + (size_t)(ng*64 + nt2*16 + nl)*256 + k0, half), acc[nt2]);
  }
#pragma unroll
  for (int nt2 = 0; nt2 < 4; ++nt2)
#pragma unroll
    for (int r = 0; r < 8; ++r){
      int row = m0 + r + 8*half, co = ng*64 + nt2*16 + nl;
      int b = row >> 15, rem = row & 32767;
      int z = rem >> 10, y = (rem >> 5) & 31, xx = rem & 31;
      size_t oi = (((size_t)(b*CC + co)*SD + z)*SD + y)*SD + xx;
      outp[oi] = skip[oi] + b8[co] + acc[nt2][r];
    }
}

// ----------------------------------------------------------------- launcher
extern "C" void kernel_launch(void* const* d_in, const int* in_sizes, int n_in,
                              void* d_out, int out_size, void* d_ws, size_t ws_size,
                              hipStream_t stream) {
  const float* x     = (const float*)d_in[0];
  const float* qw    = (const float*)d_in[1];
  const float* Ew    = (const float*)d_in[2];
  const float* Eb    = (const float*)d_in[3];
  const float* opw   = (const float*)d_in[4];
  const float* opb   = (const float*)d_in[5];
  const float* op2w  = (const float*)d_in[6];
  const float* op2b  = (const float*)d_in[7];
  const float* lnw   = (const float*)d_in[8];
  const float* lnb   = (const float*)d_in[9];
  const float* gamma = (const float*)d_in[10];
  const float* t1    = (const float*)d_in[11];
  const float* t2    = (const float*)d_in[12];
  const float* c1w   = (const float*)d_in[13];
  const float* c2w   = (const float*)d_in[14];
  const float* c8w   = (const float*)d_in[15];
  const float* c8b   = (const float*)d_in[16];
  float* outp = (float*)d_out;

  char* ws = (char*)d_ws;
  const size_t MB = 1024ull*1024ull;
  bf16_t* qkvv     = (bf16_t*)(ws);             // 128MB, dead after k_attn
  float*  conv_out = (float*)(ws);              // 64MB, reuses qkvv region
  bf16_t* xn   = (bf16_t*)(ws + 128*MB);        // 32MB
  bf16_t* act0 = xn;                            // reuse (xn dead after QKVV GEMM)
  bf16_t* xca  = (bf16_t*)(ws + 160*MB);        // 32MB
  bf16_t* act1 = xca;                           // reuse
  bf16_t* xsa  = (bf16_t*)(ws + 192*MB);        // 32MB
  bf16_t* act2 = xsa;                           // reuse
  float*  askip = (float*)(ws + 224*MB);        // 64MB fp32, long-lived
  char* wp = ws + 288*MB;
  bf16_t* qw_t = (bf16_t*)wp;  wp += (size_t)1024*256*2;
  bf16_t* EwT  = (bf16_t*)wp;  wp += (size_t)64*1024*2;
  bf16_t* W1t  = (bf16_t*)wp;  wp += (size_t)128*256*2;
  bf16_t* W2t  = (bf16_t*)wp;  wp += (size_t)128*256*2;
  bf16_t* c1t  = (bf16_t*)wp;  wp += (size_t)27*256*256*2;
  bf16_t* c2t  = (bf16_t*)wp;  wp += (size_t)27*256*256*2;
  bf16_t* c8t  = (bf16_t*)wp;  wp += (size_t)256*256*2;
  float*  inv  = (float*)wp;   wp += (size_t)64*512*4;
  float*  stats= (float*)wp;

  k_ln<<<2048, 256, 0, stream>>>(x, lnw, lnb, xn);
  k_castT<<<(256*1024+255)/256, 256, 0, stream>>>(qw,  qw_t, 256, 1024);
  k_castT<<<(1024*64+255)/256,  256, 0, stream>>>(Ew,  EwT, 1024, 64);
  k_castT<<<(256*128+255)/256,  256, 0, stream>>>(opw,  W1t, 256, 128);
  k_castT<<<(256*128+255)/256,  256, 0, stream>>>(op2w, W2t, 256, 128);
  k_castConvW<<<(256*256*27+255)/256, 256, 0, stream>>>(c1w, c1t);
  k_castConvW<<<(256*256*27+255)/256, 256, 0, stream>>>(c2w, c2t);
  k_cast<<<(256*256+255)/256, 256, 0, stream>>>(c8w, c8t, 256*256);

  k_gemm_qkvv<<<8192, 256, 0, stream>>>(xn, qw_t, qkvv);
  k_zero<<<(64*512+255)/256, 256, 0, stream>>>(inv, 64*512);
  k_invpart<<<1024, 256, 0, stream>>>(qkvv, inv);
  k_invfin<<<(64*512+255)/256, 256, 0, stream>>>(inv, 64*512);
  k_attn<<<512, 128, 0, stream>>>(qkvv, EwT, Eb, inv, t1, t2, xca, xsa);
  k_proj<<<1024, 256, 0, stream>>>(xsa, xca, W1t, W2t, opb, op2b, gamma, x, askip, act0);

  k_conv<<<2048, 256, 0, stream>>>(act0, c1t, conv_out);
  k_instat<<<512, 256, 0, stream>>>(conv_out, stats);
  k_inapply<<<65536, 256, 0, stream>>>(conv_out, stats, askip, act1, 1);
  k_conv<<<2048, 256, 0, stream>>>(act1, c2t, conv_out);
  k_instat<<<512, 256, 0, stream>>>(conv_out, stats);
  k_inapply<<<65536, 256, 0, stream>>>(conv_out, stats, askip, act2, 2);
  k_conv8<<<2048, 256, 0, stream>>>(act2, c8t, c8b, askip, outp);

  (void)in_sizes; (void)n_in; (void)out_size; (void)ws_size;
}
